// dynedge_23029614641723
// MI455X (gfx1250) — compile-verified, matrix operand order checked by reading, not run on hardware
//
#include <hip/hip_runtime.h>
#include <hip/hip_bf16.h>

typedef __bf16 bf16;
typedef __attribute__((ext_vector_type(16))) __bf16 v16bf;
typedef __attribute__((ext_vector_type(8)))  __bf16 v8bf;
typedef __attribute__((ext_vector_type(4)))  __bf16 v4bf;
typedef __attribute__((ext_vector_type(8)))  float  v8f;

#define LRELU_SLOPE 0.01f

__device__ __forceinline__ float lrelu(float v) { return v > 0.0f ? v : LRELU_SLOPE * v; }

__device__ __forceinline__ bf16 tobf(float f) {
    unsigned u = __builtin_bit_cast(unsigned, f);
    unsigned r = u + 0x7FFFu + ((u >> 16) & 1u);   // round-to-nearest-even
    unsigned short s = (unsigned short)(r >> 16);
    return __builtin_bit_cast(bf16, s);
}

__device__ __forceinline__ v8bf pack8(float4 a, float4 b) {
    v8bf r;
    r[0] = tobf(a.x); r[1] = tobf(a.y); r[2] = tobf(a.z); r[3] = tobf(a.w);
    r[4] = tobf(b.x); r[5] = tobf(b.y); r[6] = tobf(b.z); r[7] = tobf(b.w);
    return r;
}

// Build one 16-elem bf16 WMMA fragment from K-contiguous LDS row.
// Lanes 0-15 hold K 0-7 / 16-23, lanes 16-31 hold K 8-15 / 24-31 (p already
// offset by half*8): two ds_load_b128.
__device__ __forceinline__ v16bf load_frag(const bf16* p) {
    v8bf lo = *(const v8bf*)p;
    v8bf hi = *(const v8bf*)(p + 16);
    v16bf f;
    #pragma unroll
    for (int e = 0; e < 8; ++e) { f[e] = lo[e]; f[e + 8] = hi[e]; }
    return f;
}

// ---------------------------------------------------------------------------
// Weight prep: f32 (K x N) -> bf16 W^T (Npad x Kpad), zero-padded.
// ---------------------------------------------------------------------------
__global__ void transpose_bf16_kernel(const float* __restrict__ src, bf16* __restrict__ dst,
                                      int K, int N, int Kpad, int Npad) {
    int id = blockIdx.x * 256 + threadIdx.x;
    if (id < Npad * Kpad) {
        int n = id / Kpad, k = id - n * Kpad;
        float v = (n < N && k < K) ? src[(size_t)k * N + n] : 0.0f;
        dst[id] = tobf(v);
    }
}

// Combined conv first-layer weight: Wc = [w1_top - w1_bot | w1_bot],
// stored as W^T bf16, shape (Npad x Fpad), zero-padded.  w1 is (2F x L) f32.
__global__ void combine_w1_kernel(const float* __restrict__ w1, bf16* __restrict__ dst,
                                  int F, int L, int Fpad, int Npad) {
    int id = blockIdx.x * 256 + threadIdx.x;
    if (id < Npad * Fpad) {
        int n = id / Fpad, k = id - n * Fpad;
        float v = 0.0f;
        if (k < F && n < 2 * L) {
            if (n < L) v = w1[(size_t)k * L + n] - w1[(size_t)(k + F) * L + n];
            else       v = w1[(size_t)(k + F) * L + (n - L)];
        }
        dst[id] = tobf(v);
    }
}

// x (B*N,4) -> XP (B*N,32) zero-padded (for the K=32 conv1 pointwise GEMM)
__global__ void xpad_kernel(const float* __restrict__ x, float* __restrict__ xp) {
    int id = blockIdx.x * 256 + threadIdx.x;     // < BN*32
    int row = id >> 5, c = id & 31;
    xp[id] = (c < 4) ? x[row * 4 + c] : 0.0f;
}

// x into CAT cols 0..3 and zero CAT pad cols 772..799 (stride 800)
__global__ void cat_init_kernel(const float* __restrict__ x, float* __restrict__ cat) {
    int id = blockIdx.x * 256 + threadIdx.x;     // < BN*32
    int row = id >> 5, c = id & 31;
    if (c < 4)       cat[(size_t)row * 800 + c] = x[row * 4 + c];
    else             cat[(size_t)row * 800 + 768 + c] = 0.0f;   // cols 772..799
}

// ---------------------------------------------------------------------------
// kNN (K=4) on first 3 features, one block (512 threads) per batch. Exact f32.
// ---------------------------------------------------------------------------
__global__ void knn_kernel(const float* __restrict__ src, int lda, int* __restrict__ idx) {
    __shared__ float ps[512][3];
    int b = blockIdx.x, i = threadIdx.x;
    const float* p = src + (size_t)(b * 512 + i) * lda;
    ps[i][0] = p[0]; ps[i][1] = p[1]; ps[i][2] = p[2];
    __syncthreads();
    float x0 = ps[i][0], x1 = ps[i][1], x2 = ps[i][2];
    float bd[4] = {3.4e38f, 3.4e38f, 3.4e38f, 3.4e38f};
    int   bi[4] = {0, 0, 0, 0};
    for (int j = 0; j < 512; ++j) {
        float d0 = x0 - ps[j][0], d1 = x1 - ps[j][1], d2c = x2 - ps[j][2];
        float d2 = d0 * d0 + d1 * d1 + d2c * d2c;
        d2 = (j == i) ? 1.0e10f : d2;           // SELF_EXCLUDE, branch-free
        if (d2 < bd[3]) {                       // strict < keeps lower index on ties
            bd[3] = d2; bi[3] = j;
            if (bd[3] < bd[2]) { float t = bd[2]; bd[2] = bd[3]; bd[3] = t; int u = bi[2]; bi[2] = bi[3]; bi[3] = u;
            if (bd[2] < bd[1]) { t = bd[1]; bd[1] = bd[2]; bd[2] = t; u = bi[1]; bi[1] = bi[2]; bi[2] = u;
            if (bd[1] < bd[0]) { t = bd[0]; bd[0] = bd[1]; bd[1] = t; u = bi[0]; bi[0] = bi[1]; bi[1] = u; } } }
        }
    }
    int base = (b * 512 + i) * 4;
    idx[base + 0] = bi[0]; idx[base + 1] = bi[1]; idx[base + 2] = bi[2]; idx[base + 3] = bi[3];
}

// ---------------------------------------------------------------------------
// Edge build: h1[point i, edge j, :] = leaky(p[i] + q[nbr] + b1), bf16,
// zero-padded to Lpad.  One block per point (4 edges share p-row).
// PQ row = [p (L) | q (L) | pad], stride 512.  blockDim = Lpad.
// ---------------------------------------------------------------------------
__global__ void edge_kernel(const float* __restrict__ PQ, const int* __restrict__ idx,
                            const float* __restrict__ b1, bf16* __restrict__ E,
                            int L, int Lpad) {
    int i = blockIdx.x;                          // global point (b*512 + ii)
    int t = threadIdx.x;                         // < Lpad
    int cols = Lpad >> 2;
    int j = t / cols;
    int c = (t - j * cols) * 4;
    int row = i * 4 + j;                         // global edge row
    int nbr = idx[row];                          // within-batch neighbor
    int gn  = (i & ~511) + nbr;                  // global neighbor point
    bool valid = c < L;
    float4 pv = {0, 0, 0, 0}, qv = {0, 0, 0, 0}, bb = {0, 0, 0, 0};
    if (valid) {
        pv = *(const float4*)&PQ[(size_t)i * 512 + c];
        qv = *(const float4*)&PQ[(size_t)gn * 512 + L + c];
        bb = *(const float4*)&b1[c];
    }
    v4bf o;
    o[0] = tobf(valid ? lrelu(pv.x + qv.x + bb.x) : 0.0f);
    o[1] = tobf(valid ? lrelu(pv.y + qv.y + bb.y) : 0.0f);
    o[2] = tobf(valid ? lrelu(pv.z + qv.z + bb.z) : 0.0f);
    o[3] = tobf(valid ? lrelu(pv.w + qv.w + bb.w) : 0.0f);
    *(v4bf*)&E[(size_t)row * Lpad + c] = o;
}

// ---------------------------------------------------------------------------
// Tiled WMMA GEMM: out = act(A @ W + bias) [optional fused sum over 4 rows].
// Block: 256 threads = 8 waves; tile 128(M) x 64(N); K stepped by 32.
// Wave (wM 0..3, wN 0..1) computes 32x32 via 4x v_wmma_f32_16x16x32_bf16.
// All K are multiples of 32 and WT is (Npad x K) zero-padded -> branch-free
// staging: pure b128 global loads + b128 LDS stores.
// ---------------------------------------------------------------------------
template <bool ABF16, bool BIAS, bool ACT, bool RED>
__global__ __launch_bounds__(256)
void gemm_wmma(const void* __restrict__ Ap, int lda,
               const bf16* __restrict__ WT, const float* __restrict__ bias,
               float* __restrict__ out, int Nc, int K, int ostride, int ocol) {
    __shared__ bf16 ldsA[128][32];   // 8 KB
    __shared__ bf16 ldsB[64][32];    // 4 KB

    const int tid  = threadIdx.x;
    const int lane = tid & 31, wave = tid >> 5;
    const int wM = wave >> 1, wN = wave & 1;
    const int m15 = lane & 15, half = lane >> 4;

    const int mBase = blockIdx.x * 128;
    const int nTile = blockIdx.y * 64;

    const float* Af = (const float*)Ap;
    const bf16*  Ab = (const bf16*)Ap;

    // staging assignment (branch-free)
    const int arow = tid >> 1, ak = (tid & 1) * 16;   // 16 A elems / thread
    const int brow = tid >> 2, bk = (tid & 3) * 8;    //  8 B elems / thread

    v8f acc[2][2] = {};
    const int ksteps = K >> 5;

    for (int kk = 0; kk < ksteps; ++kk) {
        const int kBase = kk << 5;
        if (ABF16) {
            const bf16* s = Ab + (size_t)(mBase + arow) * lda + kBase + ak;
            *(v8bf*)&ldsA[arow][ak]     = *(const v8bf*)s;
            *(v8bf*)&ldsA[arow][ak + 8] = *(const v8bf*)(s + 8);
        } else {
            const float4* s = (const float4*)(Af + (size_t)(mBase + arow) * lda + kBase + ak);
            float4 f0 = s[0], f1 = s[1], f2 = s[2], f3 = s[3];
            *(v8bf*)&ldsA[arow][ak]     = pack8(f0, f1);
            *(v8bf*)&ldsA[arow][ak + 8] = pack8(f2, f3);
        }
        *(v8bf*)&ldsB[brow][bk] =
            *(const v8bf*)(WT + (size_t)(nTile + brow) * K + kBase + bk);
        __syncthreads();

        v16bf a0 = load_frag(&ldsA[wM * 32 + m15][half * 8]);
        v16bf a1 = load_frag(&ldsA[wM * 32 + 16 + m15][half * 8]);
        v16bf b0 = load_frag(&ldsB[wN * 32 + m15][half * 8]);
        v16bf b1 = load_frag(&ldsB[wN * 32 + 16 + m15][half * 8]);
        acc[0][0] = __builtin_amdgcn_wmma_f32_16x16x32_bf16(false, a0, false, b0, (short)0, acc[0][0], false, false);
        acc[0][1] = __builtin_amdgcn_wmma_f32_16x16x32_bf16(false, a0, false, b1, (short)0, acc[0][1], false, false);
        acc[1][0] = __builtin_amdgcn_wmma_f32_16x16x32_bf16(false, a1, false, b0, (short)0, acc[1][0], false, false);
        acc[1][1] = __builtin_amdgcn_wmma_f32_16x16x32_bf16(false, a1, false, b1, (short)0, acc[1][1], false, false);
        __syncthreads();
    }

    // epilogue; C layout: VGPR r = row r (lanes 0-15) / row r+8 (lanes 16-31)
    #pragma unroll
    for (int i = 0; i < 2; ++i) {
        #pragma unroll
        for (int j = 0; j < 2; ++j) {
            const v8f& a = acc[i][j];
            int col = nTile + wN * 32 + j * 16 + m15;
            bool cok = col < Nc;
            float bv = 0.0f;
            if (BIAS) bv = cok ? bias[col] : 0.0f;
            if (RED) {
                // 16 rows = 4 points x 4 edges: fused leaky + sum over K=4
                float t0 = lrelu(a[0] + bv), t1 = lrelu(a[1] + bv);
                float t2 = lrelu(a[2] + bv), t3 = lrelu(a[3] + bv);
                float t4 = lrelu(a[4] + bv), t5 = lrelu(a[5] + bv);
                float t6 = lrelu(a[6] + bv), t7 = lrelu(a[7] + bv);
                float s0 = t0 + t1 + t2 + t3;
                float s1 = t4 + t5 + t6 + t7;
                int p0 = ((mBase + wM * 32 + i * 16) >> 2) + half * 2;
                out[(size_t)p0 * ostride + ocol + col]       = cok ? s0 : 0.0f;
                out[(size_t)(p0 + 1) * ostride + ocol + col] = cok ? s1 : 0.0f;
            } else {
                int rowBase = mBase + wM * 32 + i * 16 + half * 8;
                #pragma unroll
                for (int r = 0; r < 8; ++r) {
                    float v = a[r] + bv;
                    if (ACT) v = lrelu(v);
                    // unconditional store: dest strides are padded, pad cols get 0
                    out[(size_t)(rowBase + r) * ostride + ocol + col] = cok ? v : 0.0f;
                }
            }
        }
    }
}

// ---------------------------------------------------------------------------
// Head: per-batch max/min/sum/mean pooling over N=512, leaky, nn3, leaky, nn4.
// ---------------------------------------------------------------------------
__global__ void head_kernel(const float* __restrict__ H,
                            const float* __restrict__ w3, const float* __restrict__ b3,
                            const float* __restrict__ w4, const float* __restrict__ b4,
                            float* __restrict__ out) {
    __shared__ float pooled[768];
    __shared__ float g[96];
    int b = blockIdx.x, t = threadIdx.x;
    if (t < 192) {
        float mx = -3.4e38f, mn = 3.4e38f, sm = 0.0f;
        for (int n = 0; n < 512; ++n) {
            float v = H[((size_t)b * 512 + n) * 192 + t];
            mx = fmaxf(mx, v); mn = fminf(mn, v); sm += v;
        }
        pooled[t]       = lrelu(mx);
        pooled[192 + t] = lrelu(mn);
        pooled[384 + t] = lrelu(sm);
        pooled[576 + t] = lrelu(sm * (1.0f / 512.0f));
    }
    __syncthreads();
    if (t < 96) {
        float a = b3[t];
        for (int k = 0; k < 768; ++k) a += pooled[k] * w3[(size_t)k * 96 + t];
        g[t] = lrelu(a);
    }
    __syncthreads();
    if (t == 0) {
        float a = b4[0];
        for (int k = 0; k < 96; ++k) a += g[k] * w4[k];
        out[b] = a;
    }
}

// ---------------------------------------------------------------------------
extern "C" void kernel_launch(void* const* d_in, const int* in_sizes, int n_in,
                              void* d_out, int out_size, void* d_ws, size_t ws_size,
                              hipStream_t stream) {
    (void)in_sizes; (void)n_in; (void)out_size; (void)ws_size;

    const float* x      = (const float*)d_in[0];
    const float* cw1[4] = {(const float*)d_in[1],  (const float*)d_in[5],
                           (const float*)d_in[9],  (const float*)d_in[13]};
    const float* cb1[4] = {(const float*)d_in[2],  (const float*)d_in[6],
                           (const float*)d_in[10], (const float*)d_in[14]};
    const float* cw2[4] = {(const float*)d_in[3],  (const float*)d_in[7],
                           (const float*)d_in[11], (const float*)d_in[15]};
    const float* cb2[4] = {(const float*)d_in[4],  (const float*)d_in[8],
                           (const float*)d_in[12], (const float*)d_in[16]};
    const float* nn1_w = (const float*)d_in[17]; const float* nn1_b = (const float*)d_in[18];
    const float* nn2_w = (const float*)d_in[19]; const float* nn2_b = (const float*)d_in[20];
    const float* nn3_w = (const float*)d_in[21]; const float* nn3_b = (const float*)d_in[22];
    const float* nn4_w = (const float*)d_in[23]; const float* nn4_b = (const float*)d_in[24];
    float* out = (float*)d_out;

    const int B = 128, N = 512;
    const int BN = B * N;              // 65536
    const int ME = BN * 4;             // 262144 edge rows

    // workspace layout (all padded strides)
    char* base = (char*)d_ws;
    size_t off = 0;
    auto alloc = [&](size_t bytes) { size_t o = off; off = (off + bytes + 255) & ~(size_t)255; return o; };
    size_t o_wc1 = alloc((size_t)2 * 192 * 32);
    size_t o_wc[3];
    for (int c = 0; c < 3; ++c) o_wc[c] = alloc((size_t)2 * 512 * 192);
    size_t o_w21 = alloc((size_t)2 * 192 * 96);
    size_t o_w2[3];
    for (int c = 0; c < 3; ++c) o_w2[c] = alloc((size_t)2 * 192 * 256);
    size_t o_nn1 = alloc((size_t)2 * 256 * 800);
    size_t o_nn2 = alloc((size_t)2 * 192 * 256);
    size_t o_idx = alloc((size_t)ME * 4);
    size_t o_xp  = alloc((size_t)BN * 32 * 4);
    size_t o_cat = alloc((size_t)BN * 800 * 4);
    size_t o_pq  = alloc((size_t)BN * 512 * 4);    // reused for G1 (BN x 256 f32)
    size_t o_e   = alloc((size_t)ME * 256 * 2);    // reused for H  (BN x 192 f32)

    bf16* wc1T   = (bf16*)(base + o_wc1);
    bf16* wcT[3] = {(bf16*)(base + o_wc[0]), (bf16*)(base + o_wc[1]), (bf16*)(base + o_wc[2])};
    bf16* w21T   = (bf16*)(base + o_w21);
    bf16* w2T[3] = {(bf16*)(base + o_w2[0]), (bf16*)(base + o_w2[1]), (bf16*)(base + o_w2[2])};
    bf16* nn1T   = (bf16*)(base + o_nn1);
    bf16* nn2T   = (bf16*)(base + o_nn2);
    int*   idxb  = (int*)(base + o_idx);
    float* XP    = (float*)(base + o_xp);
    float* CAT   = (float*)(base + o_cat);
    float* PQ    = (float*)(base + o_pq);
    bf16*  E     = (bf16*)(base + o_e);
    float* G1    = PQ;                   // reuse
    float* H     = (float*)(base + o_e); // reuse

    auto blocks = [](int n) { return (n + 255) / 256; };

    // weight prep (padded, zero-filled)
    combine_w1_kernel<<<blocks(192 * 32), 256, 0, stream>>>(cw1[0], wc1T, 4, 96, 32, 192);
    for (int c = 0; c < 3; ++c)
        combine_w1_kernel<<<blocks(512 * 192), 256, 0, stream>>>(cw1[c + 1], wcT[c], 192, 252, 192, 512);
    transpose_bf16_kernel<<<blocks(192 * 96), 256, 0, stream>>>(cw2[0], w21T, 96, 192, 96, 192);
    for (int c = 0; c < 3; ++c)
        transpose_bf16_kernel<<<blocks(192 * 256), 256, 0, stream>>>(cw2[c + 1], w2T[c], 252, 192, 256, 192);
    transpose_bf16_kernel<<<blocks(256 * 800), 256, 0, stream>>>(nn1_w, nn1T, 772, 252, 800, 256);
    transpose_bf16_kernel<<<blocks(192 * 256), 256, 0, stream>>>(nn2_w, nn2T, 252, 192, 256, 192);

    xpad_kernel<<<blocks(BN * 32), 256, 0, stream>>>(x, XP);
    cat_init_kernel<<<blocks(BN * 32), 256, 0, stream>>>(x, CAT);

    // four EdgeConv blocks (dynamic graph: kNN on first 3 features of input)
    for (int c = 0; c < 4; ++c) {
        const float* src  = (c == 0) ? XP : (CAT + 4 + (size_t)(c - 1) * 192);
        int lda  = (c == 0) ? 32 : 800;
        int Kp   = (c == 0) ? 32 : 192;     // pointwise K (padded)
        int L    = (c == 0) ? 96 : 252;
        int Lpad = (c == 0) ? 96 : 256;
        int twoL = 2 * L;
        int nTiles = (c == 0) ? 3 : 8;      // 192 / 504->512 cols
        bf16* wc = (c == 0) ? wc1T : wcT[c - 1];
        bf16* w2 = (c == 0) ? w21T : w2T[c - 1];

        knn_kernel<<<B, 512, 0, stream>>>(src, lda, idxb);

        // pointwise: PQ = src @ [w1t-w1b | w1b]   (M=BN, K=Kp, N=2L, stride 512)
        gemm_wmma<false, false, false, false>
            <<<dim3(BN / 128, nTiles), 256, 0, stream>>>(
                src, lda, wc, nullptr, PQ, twoL, Kp, 512, 0);

        // gather + bias + leaky -> bf16 edge rows (padded to Lpad)
        edge_kernel<<<BN, Lpad, 0, stream>>>(PQ, idxb, cb1[c], E, L, Lpad);

        // h2 = leaky(E @ w2 + b2), fused sum over 4 neighbor rows -> CAT slice
        gemm_wmma<true, true, true, true>
            <<<dim3(ME / 128, 3), 256, 0, stream>>>(
                E, Lpad, w2, cb2[c], CAT, 192, Lpad, 800, 4 + c * 192);
    }

    // nn1: leaky(CAT @ nn1_w + b)   (BN x 800 -> 252, out stride 256 zero-padded)
    gemm_wmma<false, true, true, false>
        <<<dim3(BN / 128, 4), 256, 0, stream>>>(CAT, 800, nn1T, nn1_b, G1, 252, 800, 256, 0);
    // nn2: G1 @ nn2_w + b           (BN x 256 -> 192)
    gemm_wmma<false, true, false, false>
        <<<dim3(BN / 128, 3), 256, 0, stream>>>(G1, 256, nn2T, nn2_b, H, 192, 256, 192, 0);

    // pooling + nn3 + nn4
    head_kernel<<<B, 256, 0, stream>>>(H, nn3_w, nn3_b, nn4_w, nn4_b, out);
}
